// StackedLSTMCell_75385265979695
// MI455X (gfx1250) — matrix operand order repaired and checked by reading, not compiled, not run on hardware
//
#include <hip/hip_runtime.h>

// ---------------------------------------------------------------------------
// Stacked LSTM cell (64 models) on gfx1250 — split-bf16 WMMA GEMM.
//
// Roofline: 17.2 GFLOP over ~560 MB HBB traffic -> ~24 us memory bound needs
// ~0.72 PFLOP/s. f32 WMMA (16x16x4, 2048 FLOP/instr) is ~2-3x short; instead
// each f32 operand is split v = hi + lo (two RNE bf16, ~16 mantissa bits) and
// z += Ah*Bh + Ah*Bl + Al*Bh with V_WMMA_F32_16X16X32_BF16 (f32 accumulate):
// 3 WMMA per K=32 instead of 8, error ~2^-18 (f32-grade for K=1024 dots).
//
// One block per (model, 32-unit column group); 8 wave32; wave (mt,cth) owns a
// 16x16 tile for ALL four gates (4 x v8f acc) so the LSTM nonlinearity fuses
// in registers. A staged as [row][k] bf16 hi/lo; B staged TRANSPOSED
// [col][k] bf16 hi/lo (16-bit WMMA frags are K-major per lane). Fragments are
// aligned ds_load_b128 pairs; staging global loads are coalesced.
// ---------------------------------------------------------------------------

typedef __attribute__((ext_vector_type(8)))  float  v8f;
typedef __attribute__((ext_vector_type(8)))  __bf16 v8bf;
typedef __attribute__((ext_vector_type(16))) __bf16 v16bf;

#define MODELS    64
#define BATCH     64
#define INPUT_DIM 512
#define UNITS     512
#define KDIM      1024   // INPUT_DIM + UNITS
#define ZW        2048   // 4*UNITS
#define KC        64     // K chunk staged per iteration

// LDS row strides in bf16 elements: 64 K + 8 pad = 144B rows (16B aligned);
// row*144B/4 mod 64 tiles all banks across a half-wave on b128 frag reads.
#define AS_S 72
#define BS_S 72

__device__ __forceinline__ float sigmoidf_(float x) {
  return 1.0f / (1.0f + __expf(-x));
}
__device__ __forceinline__ float tanhf_(float x) {
  float e = __expf(2.0f * x);
  return (e - 1.0f) / (e + 1.0f);
}

#define WMMA_BF16(a, b, c) \
  __builtin_amdgcn_wmma_f32_16x16x32_bf16(false, (a), false, (b), (short)0, (c), false, false)

// A frag (16x32 bf16): lane half hif -> 16 consecutive K at [16*hif)
__device__ __forceinline__ v16bf ldfragA(const __bf16* p) {
  v8bf a = *(const v8bf*)p;
  v8bf b = *(const v8bf*)(p + 8);
  return __builtin_shufflevector(a, b, 0,1,2,3,4,5,6,7,8,9,10,11,12,13,14,15);
}
// B frag (32x16 bf16): lane half hif -> 16 consecutive K at [16*hif)
__device__ __forceinline__ v16bf ldfragB(const __bf16* p) {
  v8bf a = *(const v8bf*)p;
  v8bf b = *(const v8bf*)(p + 8);
  return __builtin_shufflevector(a, b, 0,1,2,3,4,5,6,7,8,9,10,11,12,13,14,15);
}

__global__ void __launch_bounds__(256)
lstm_wmma_bf16split(const float* __restrict__ x,   // [64,64,512]
                    const float* __restrict__ h,   // [64,64,512]
                    const float* __restrict__ cpr, // [64,64,512]
                    const float* __restrict__ wk,  // [64,512,2048]
                    const float* __restrict__ wr,  // [64,512,2048]
                    const float* __restrict__ bia, // [64,1,2048]
                    float* __restrict__ out)       // h | h | c
{
  __shared__ __align__(16) __bf16 AsH[BATCH * AS_S];  // 64 rows x 64 K (hi)
  __shared__ __align__(16) __bf16 AsL[BATCH * AS_S];  // 64 rows x 64 K (lo)
  __shared__ __align__(16) __bf16 BsH[128 * BS_S];    // 128 cols x 64 K (hi), transposed
  __shared__ __align__(16) __bf16 BsL[128 * BS_S];    // 128 cols x 64 K (lo), transposed

  const int tid  = threadIdx.x;
  const int lane = tid & 31;
  const int wave = tid >> 5;
  const int hif  = lane >> 4;        // wave half: 0 = lanes 0-15, 1 = 16-31
  const int l15  = lane & 15;
  const int mt   = wave & 3;         // M tile (rows mt*16..+16)
  const int cth  = wave >> 2;        // column-tile half within the 32-unit group
  const int m    = blockIdx.x >> 4;  // model
  const int cg   = blockIdx.x & 15;  // 32-unit column group
  const int u0   = cg * 32 + cth * 16;

  const float* xm  = x  + (size_t)m * BATCH * INPUT_DIM;
  const float* hm  = h  + (size_t)m * BATCH * UNITS;
  const float* wkm = wk + (size_t)m * INPUT_DIM * ZW;
  const float* wrm = wr + (size_t)m * UNITS * ZW;

  v8f acc0 = {}, acc1 = {}, acc2 = {}, acc3 = {};   // i, f, g, o gate tiles

  for (int kb = 0; kb < KDIM; kb += KC) {
    const bool lowK = (kb < INPUT_DIM);
    const float* Asrc = lowK ? xm : hm;
    const float* Bsrc = lowK ? wkm : wrm;
    const int kl = kb & (INPUT_DIM - 1);

    // ---- stage A: 64 rows x 64 K -> bf16 hi/lo, [row][k] ----
    {
      const int row = tid >> 2;              // 0..63
      const int ks  = (tid & 3) << 4;        // 0,16,32,48
      const float* src = Asrc + (size_t)row * INPUT_DIM + kl + ks;
      v8bf h0, h1, l0, l1;
#pragma unroll
      for (int i = 0; i < 16; ++i) {
        const float  f  = src[i];            // backend merges into b128 loads
        const __bf16 hb = (__bf16)f;
        const __bf16 lb = (__bf16)(f - (float)hb);
        if (i < 8) { h0[i] = hb; l0[i] = lb; }
        else       { h1[i - 8] = hb; l1[i - 8] = lb; }
      }
      __bf16* dh = &AsH[row * AS_S + ks];
      __bf16* dl = &AsL[row * AS_S + ks];
      *(v8bf*)dh = h0;  *(v8bf*)(dh + 8) = h1;
      *(v8bf*)dl = l0;  *(v8bf*)(dl + 8) = l1;
    }
    // ---- stage B transposed: 64 K x 128 cols -> bf16 hi/lo, [col][k] ----
    // thread = (unit col c4, K group kg); one column per gate; every dword
    // load below is a fully coalesced 128B wave request (lanes = c4).
    {
      const int c4 = tid & 31;               // unit col within group
      const int kg = tid >> 5;               // K group 0..7 (8 rows each)
      const float* src = Bsrc + (size_t)(kl + kg * 8) * ZW + cg * 32 + c4;
#pragma unroll
      for (int j = 0; j < 4; ++j) {          // gate
        v8bf hb, lb;
#pragma unroll
        for (int r = 0; r < 8; ++r) {
          const float  f = src[(size_t)r * ZW + j * UNITS];
          const __bf16 hv = (__bf16)f;
          hb[r] = hv;
          lb[r] = (__bf16)(f - (float)hv);
        }
        const int col = j * 32 + c4;
        *(v8bf*)&BsH[col * BS_S + kg * 8] = hb;
        *(v8bf*)&BsL[col * BS_S + kg * 8] = lb;
      }
    }

    // ---- L2 prefetch of next chunk (global_prefetch_b8) ----
    if (kb + KC < KDIM) {
      const int  kn   = kb + KC;
      const bool lowN = (kn < INPUT_DIM);
      const int  knl  = kn & (INPUT_DIM - 1);
      const float* Bn = lowN ? wkm : wrm;
      const float* An = lowN ? xm : hm;
      {  // 64 K rows x 4 gate segments, one 128B line each
        const int row = tid & 63;
        const int g   = tid >> 6;
        __builtin_prefetch(Bn + (size_t)(knl + row) * ZW + g * UNITS + cg * 32, 0, 1);
      }
      if (tid < 128) {  // 64 A rows x 2 lines
        __builtin_prefetch(An + (size_t)(tid >> 1) * INPUT_DIM + knl + (tid & 1) * 32, 0, 1);
      }
    }
    __syncthreads();

    // ---- compute: 2 K-steps of 32; per gate 3 WMMA (AhBh, AhBl, AlBh) ----
    const __bf16* aHp = &AsH[(mt * 16 + l15) * AS_S + 16 * hif];
    const __bf16* aLp = &AsL[(mt * 16 + l15) * AS_S + 16 * hif];
    const __bf16* bHp = &BsH[(cth * 16 + l15) * BS_S + 16 * hif];
    const __bf16* bLp = &BsL[(cth * 16 + l15) * BS_S + 16 * hif];
#pragma unroll
    for (int kk = 0; kk < KC; kk += 32) {
      const v16bf aH = ldfragA(aHp + kk);
      const v16bf aL = ldfragA(aLp + kk);
      {
        const v16bf bH = ldfragB(bHp + kk + 0 * 32 * BS_S);
        const v16bf bL = ldfragB(bLp + kk + 0 * 32 * BS_S);
        acc0 = WMMA_BF16(aH, bH, acc0);
        acc0 = WMMA_BF16(aH, bL, acc0);
        acc0 = WMMA_BF16(aL, bH, acc0);
      }
      {
        const v16bf bH = ldfragB(bHp + kk + 1 * 32 * BS_S);
        const v16bf bL = ldfragB(bLp + kk + 1 * 32 * BS_S);
        acc1 = WMMA_BF16(aH, bH, acc1);
        acc1 = WMMA_BF16(aH, bL, acc1);
        acc1 = WMMA_BF16(aL, bH, acc1);
      }
      {
        const v16bf bH = ldfragB(bHp + kk + 2 * 32 * BS_S);
        const v16bf bL = ldfragB(bLp + kk + 2 * 32 * BS_S);
        acc2 = WMMA_BF16(aH, bH, acc2);
        acc2 = WMMA_BF16(aH, bL, acc2);
        acc2 = WMMA_BF16(aL, bH, acc2);
      }
      {
        const v16bf bH = ldfragB(bHp + kk + 3 * 32 * BS_S);
        const v16bf bL = ldfragB(bLp + kk + 3 * 32 * BS_S);
        acc3 = WMMA_BF16(aH, bH, acc3);
        acc3 = WMMA_BF16(aH, bL, acc3);
        acc3 = WMMA_BF16(aL, bH, acc3);
      }
    }
    __syncthreads();
  }

  // ---- fused gate epilogue (bias added in f32, exact) ----
  const float* bm = bia + (size_t)m * ZW;
  const float bi = bm[0 * UNITS + u0 + l15];
  const float bf = bm[1 * UNITS + u0 + l15];
  const float bg = bm[2 * UNITS + u0 + l15];
  const float bo = bm[3 * UNITS + u0 + l15];

  const float* cm = cpr + (size_t)m * BATCH * UNITS;
  const size_t plane = (size_t)MODELS * BATCH * UNITS;   // 2,097,152
  const size_t mbase = (size_t)m * BATCH * UNITS;
  float* outH  = out;
  float* outH2 = out + plane;
  float* outC  = out + 2 * plane;

#pragma unroll
  for (int e = 0; e < 8; ++e) {
    const int row = mt * 16 + e + 8 * hif;        // C/D layout: vgpr e, half hif
    const int off = row * UNITS + u0 + l15;
    const float ig = sigmoidf_(acc0[e] + bi);
    const float fg = sigmoidf_(acc1[e] + bf);
    const float gg = tanhf_(acc2[e] + bg);
    const float og = sigmoidf_(acc3[e] + bo);
    const float cn = fg * cm[off] + ig * gg;
    const float hv = og * tanhf_(cn);
    outH [mbase + off] = hv;
    outH2[mbase + off] = hv;
    outC [mbase + off] = cn;
  }
}

extern "C" void kernel_launch(void* const* d_in, const int* in_sizes, int n_in,
                              void* d_out, int out_size, void* d_ws, size_t ws_size,
                              hipStream_t stream) {
  (void)in_sizes; (void)n_in; (void)out_size; (void)d_ws; (void)ws_size;
  const float* x   = (const float*)d_in[0];
  const float* h   = (const float*)d_in[1];
  const float* c   = (const float*)d_in[2];
  const float* wk  = (const float*)d_in[3];
  const float* wr  = (const float*)d_in[4];
  const float* bia = (const float*)d_in[5];
  float* out = (float*)d_out;

  dim3 grid(MODELS * 16);   // (model, 32-unit column group)
  dim3 block(256);          // 8 wave32
  hipLaunchKernelGGL(lstm_wmma_bf16split, grid, block, 0, stream,
                     x, h, c, wk, wr, bia, out);
}
